// SRUCell_1271310320379
// MI455X (gfx1250) — matrix operand order repaired and checked by reading, not compiled, not run on hardware
//
#include <hip/hip_runtime.h>

// ---------------------------------------------------------------------------
// SRU layer for MI455X (gfx1250, wave32, WMMA + TDM).
//   pass 0: fp32 -> bf16 hi/lo split of x (row-major) and W (transposed [n][k])
//   pass 1: u = x @ W via v_wmma_f32_16x16x32_bf16, 3-term split,
//           tiles staged by Tensor Data Mover (tensor_load_to_lds, TENSORcnt),
//           double-buffered LDS with hardware row padding
//   pass 2: gated linear recurrence over L (32768 independent channels)
// ---------------------------------------------------------------------------

typedef __bf16  bf16_t;
typedef bf16_t  v16bf __attribute__((ext_vector_type(16)));
typedef float   v8f   __attribute__((ext_vector_type(8)));
typedef unsigned int v4u __attribute__((ext_vector_type(4)));
typedef int          v4i __attribute__((ext_vector_type(4)));
typedef int          v8i __attribute__((ext_vector_type(8)));
typedef unsigned short ushort_t;

constexpr int Ldim = 1024;
constexpr int Bdim = 32;
constexpr int Ddim = 1024;
constexpr int Mdim = Ldim * Bdim;   // 32768
constexpr int Kdim = Ddim;          // 1024
constexpr int Ndim = 3 * Ddim;      // 3072

constexpr int BM = 128;
constexpr int BN = 128;
constexpr int BK = 32;              // == WMMA K
constexpr int RS = 40;              // LDS row stride in ushorts (32 data + 8 pad) = 80B
constexpr int NIT = Kdim / BK;      // 32 K-steps

// round-to-nearest-even float -> bf16 bits (low 16)
__device__ __forceinline__ unsigned f2bf(float f) {
    unsigned u = __float_as_uint(f);
    return (u + 0x7FFFu + ((u >> 16) & 1u)) >> 16;
}

__device__ __forceinline__ v16bf ld_frag2(const unsigned* p0, const unsigned* p1) {
    union { uint4 q[2]; v16bf v; } c;
    c.q[0] = *(const uint4*)p0;
    c.q[1] = *(const uint4*)p1;
    return c.v;
}

// ---------------- pass 0a: x -> bf16 hi/lo, row-major --------------------
__global__ __launch_bounds__(256)
void cvt_x(const float* __restrict__ X, ushort_t* __restrict__ xh, ushort_t* __restrict__ xl) {
    const int idx4 = blockIdx.x * 256 + threadIdx.x;       // one float4 per thread
    const float4 v = *(const float4*)(X + (size_t)idx4 * 4);
    float fv[4] = {v.x, v.y, v.z, v.w};
    unsigned hb[4], lb[4];
    #pragma unroll
    for (int e = 0; e < 4; ++e) {
        hb[e] = f2bf(fv[e]);
        lb[e] = f2bf(fv[e] - __uint_as_float(hb[e] << 16));
    }
    ((uint2*)xh)[idx4] = make_uint2(hb[0] | (hb[1] << 16), hb[2] | (hb[3] << 16));
    ((uint2*)xl)[idx4] = make_uint2(lb[0] | (lb[1] << 16), lb[2] | (lb[3] << 16));
}

// ---------------- pass 0b: W (K,N) -> bf16 hi/lo transposed (N,K) --------
__global__ __launch_bounds__(256)
void cvt_w(const float* __restrict__ W, ushort_t* __restrict__ wh, ushort_t* __restrict__ wl) {
    __shared__ ushort_t th[32][36];
    __shared__ ushort_t tl[32][36];
    const int tid = threadIdx.x;
    const int n0 = blockIdx.x * 32;
    const int k0 = blockIdx.y * 32;
    {
        int k  = tid >> 3;
        int n4 = tid & 7;
        const float4 v = *(const float4*)(W + (size_t)(k0 + k) * Ndim + n0 + n4 * 4);
        float fv[4] = {v.x, v.y, v.z, v.w};
        #pragma unroll
        for (int e = 0; e < 4; ++e) {
            unsigned hb = f2bf(fv[e]);
            unsigned lb = f2bf(fv[e] - __uint_as_float(hb << 16));
            th[k][n4 * 4 + e] = (ushort_t)hb;
            tl[k][n4 * 4 + e] = (ushort_t)lb;
        }
    }
    __syncthreads();
    {
        int n  = tid >> 3;
        int k4 = tid & 7;
        unsigned h0 = th[k4 * 4 + 0][n], h1 = th[k4 * 4 + 1][n];
        unsigned h2 = th[k4 * 4 + 2][n], h3 = th[k4 * 4 + 3][n];
        unsigned l0 = tl[k4 * 4 + 0][n], l1 = tl[k4 * 4 + 1][n];
        unsigned l2 = tl[k4 * 4 + 2][n], l3 = tl[k4 * 4 + 3][n];
        uint2* dh = (uint2*)(wh + (size_t)(n0 + n) * Kdim + k0);
        uint2* dl = (uint2*)(wl + (size_t)(n0 + n) * Kdim + k0);
        dh[k4] = make_uint2(h0 | (h1 << 16), h2 | (h3 << 16));
        dl[k4] = make_uint2(l0 | (l1 << 16), l2 | (l3 << 16));
    }
}

// ---------------- TDM tile load: 128 rows x 64B, stride 2048B ------------
// LDS rows padded by TDM: 4 dwords pad per 16 dwords stored -> 80B row stride.
__device__ __forceinline__ void tdm_load_tile(const ushort_t* gbase, unsigned lds_off,
                                              unsigned rows_total) {
    unsigned long long ga = (unsigned long long)(const void*)gbase;
    v4u g0 = { 1u,                                    // count=1 (valid user D#)
               lds_off,                               // LDS byte address
               (unsigned)(ga & 0xffffffffu),          // global_addr[31:0]
               (unsigned)(ga >> 32) | 0x80000000u };  // global_addr[56:32] | type=2
    v8i g1 = { (int)0x06D20000,                       // data_size=4B, pad_en,
                                                      // pad_interval=16dw, pad_amount=4dw
               (int)0x02000000,                       // tensor_dim0 = 512 dw (lo16<<16)
               (int)((rows_total & 0xffffu) << 16),   // dim0 hi=0 | tensor_dim1 lo16
               (int)((rows_total >> 16) | (16u << 16)), // tensor_dim1 hi | tile_dim0=16dw
               128,                                   // tile_dim1 = 128 rows (tile_dim2=0)
               512,                                   // tensor_dim0_stride = 512 dw (2048B)
               0, 0 };
    v4i gz4 = { 0, 0, 0, 0 };                         // groups 2/3 unused (2-D tensor)
    v8i gz8 = { 0, 0, 0, 0, 0, 0, 0, 0 };             // extra operand (clang-23 form)
    __builtin_amdgcn_tensor_load_to_lds(g0, g1, gz4, gz4, gz8, 0);
}

// ---------------- pass 1: GEMM, TDM-staged, double-buffered --------------
__global__ __launch_bounds__(256)
void sru_gemm(const ushort_t* __restrict__ xh, const ushort_t* __restrict__ xl,
              const ushort_t* __restrict__ wh, const ushort_t* __restrict__ wl,
              float* __restrict__ U) {
    __shared__ __align__(16) ushort_t Ah[2][BM * RS];
    __shared__ __align__(16) ushort_t Al[2][BM * RS];
    __shared__ __align__(16) ushort_t Bh[2][BN * RS];
    __shared__ __align__(16) ushort_t Bl[2][BN * RS];

    const int tid  = threadIdx.x;
    const int lane = tid & 31;
    const int wave = tid >> 5;
    const int wm   = wave >> 1;          // 0..3
    const int wn   = wave & 1;           // 0..1
    const int l15  = lane & 15;
    const int hsel = lane >> 4;
    const int aoff = hsel * 4;           // uint offset: A K-half select
    const int boff = hsel * 8;           // unused; B fragments use aoff (same k order as A)

    const int blockN0 = blockIdx.x * BN; // fastest-varying -> x tiles reused in L2
    const int blockM0 = blockIdx.y * BM;

    // waves 0..3 each own one TDM stream: Ah, Al, Bh, Bl
    const ushort_t* gsel = (wave == 0) ? xh : (wave == 1) ? xl : (wave == 2) ? wh : wl;
    const int       row0 = (wave < 2) ? blockM0 : blockN0;
    const unsigned  rows = (wave < 2) ? (unsigned)Mdim : (unsigned)Ndim;
    const unsigned  lds0 = (wave == 0) ? (unsigned)(size_t)&Ah[0][0]
                         : (wave == 1) ? (unsigned)(size_t)&Al[0][0]
                         : (wave == 2) ? (unsigned)(size_t)&Bh[0][0]
                         :               (unsigned)(size_t)&Bl[0][0];
    const ushort_t* gtile0 = gsel + (size_t)row0 * Kdim;   // + kk per step
    constexpr unsigned LDS_BUF_STRIDE = BM * RS * 2;       // 10240 B

    v8f acc[2][4] = {};

    if (wave < 4) tdm_load_tile(gtile0, lds0, rows);       // fill buffer 0

    for (int it = 0; it < NIT; ++it) {
        const int cur = it & 1;
        __builtin_amdgcn_s_wait_tensorcnt(0);              // buf[cur] tiles landed
        __syncthreads();
        if (wave < 4 && (it + 1) < NIT)                    // start filling buf[nxt]
            tdm_load_tile(gtile0 + (it + 1) * BK, lds0 + (cur ^ 1) * LDS_BUF_STRIDE, rows);

        v16bf ahf[2], alf[2], bhf[4], blf[4];
        #pragma unroll
        for (int mt = 0; mt < 2; ++mt) {
            const unsigned* rh = (const unsigned*)&Ah[cur][(wm * 32 + mt * 16 + l15) * RS];
            const unsigned* rl = (const unsigned*)&Al[cur][(wm * 32 + mt * 16 + l15) * RS];
            ahf[mt] = ld_frag2(rh + aoff, rh + aoff + 8);
            alf[mt] = ld_frag2(rl + aoff, rl + aoff + 8);
        }
        #pragma unroll
        for (int nt = 0; nt < 4; ++nt) {
            const unsigned* rh = (const unsigned*)&Bh[cur][(wn * 64 + nt * 16 + l15) * RS];
            const unsigned* rl = (const unsigned*)&Bl[cur][(wn * 64 + nt * 16 + l15) * RS];
            bhf[nt] = ld_frag2(rh + aoff, rh + aoff + 8);
            blf[nt] = ld_frag2(rl + aoff, rl + aoff + 8);
        }

        #pragma unroll
        for (int mt = 0; mt < 2; ++mt) {
            #pragma unroll
            for (int nt = 0; nt < 4; ++nt) {
                acc[mt][nt] = __builtin_amdgcn_wmma_f32_16x16x32_bf16(
                    false, ahf[mt], false, bhf[nt], (short)0, acc[mt][nt], false, false);
                acc[mt][nt] = __builtin_amdgcn_wmma_f32_16x16x32_bf16(
                    false, ahf[mt], false, blf[nt], (short)0, acc[mt][nt], false, false);
                acc[mt][nt] = __builtin_amdgcn_wmma_f32_16x16x32_bf16(
                    false, alf[mt], false, bhf[nt], (short)0, acc[mt][nt], false, false);
            }
        }
    }

    // store C (streamed once -> nontemporal, keep x tiles resident in L2)
    #pragma unroll
    for (int mt = 0; mt < 2; ++mt) {
        int gm = blockM0 + wm * 32 + mt * 16 + hsel * 8;
        #pragma unroll
        for (int nt = 0; nt < 4; ++nt) {
            int gn = blockN0 + wn * 64 + nt * 16 + l15;
            #pragma unroll
            for (int r = 0; r < 8; ++r)
                __builtin_nontemporal_store(acc[mt][nt][r], U + (size_t)(gm + r) * Ndim + gn);
        }
    }
}

// ---------------- pass 2: recurrence + gating ----------------------------
__global__ __launch_bounds__(256)
void sru_scan(const float* __restrict__ U, const float* __restrict__ X,
              const float* __restrict__ bias, const float* __restrict__ c0,
              float* __restrict__ H) {
    const int t = blockIdx.x * blockDim.x + threadIdx.x;  // 0 .. B*D-1
    const int b = t >> 10;
    const int d = t & (Ddim - 1);
    const float b1 = bias[d];
    const float b2 = bias[Ddim + d];
    float c = c0[t];
    for (int l = 0; l < Ldim; ++l) {
        const size_t m = (size_t)l * Bdim + b;
        const size_t ub = m * Ndim + 3 * (size_t)d;
        float u0 = U[ub + 0];
        float u1 = U[ub + 1];
        float u2 = U[ub + 2];
        float g1 = 1.0f / (1.0f + __expf(-(u1 + b1)));
        float g2 = 1.0f / (1.0f + __expf(-(u2 + b2)));
        c = (c - u0) * g1 + u0;
        float xv = X[m * Ddim + d];
        H[m * Ddim + d] = (tanhf(c) - xv) * g2 + xv;
    }
}

extern "C" void kernel_launch(void* const* d_in, const int* in_sizes, int n_in,
                              void* d_out, int out_size, void* d_ws, size_t ws_size,
                              hipStream_t stream) {
    (void)in_sizes; (void)n_in; (void)out_size; (void)ws_size;
    const float* x    = (const float*)d_in[0];
    const float* W    = (const float*)d_in[1];
    const float* bias = (const float*)d_in[2];
    const float* c0   = (const float*)d_in[3];
    float* h = (float*)d_out;

    char* p = (char*)d_ws;
    float*    u  = (float*)p;        p += (size_t)Mdim * Ndim * 4;   // 402.7 MB
    ushort_t* xh = (ushort_t*)p;     p += (size_t)Mdim * Kdim * 2;   // 67.1 MB
    ushort_t* xl = (ushort_t*)p;     p += (size_t)Mdim * Kdim * 2;
    ushort_t* wh = (ushort_t*)p;     p += (size_t)Kdim * Ndim * 2;   // 6.3 MB
    ushort_t* wl = (ushort_t*)p;

    cvt_x<<<dim3((Mdim * Kdim) / (256 * 4)), 256, 0, stream>>>(x, xh, xl);
    cvt_w<<<dim3(Ndim / 32, Kdim / 32), 256, 0, stream>>>(W, wh, wl);
    sru_gemm<<<dim3(Ndim / BN, Mdim / BM), 256, 0, stream>>>(xh, xl, wh, wl, u);
    sru_scan<<<dim3((Bdim * Ddim) / 256), 256, 0, stream>>>(u, x, bias, c0, h);
}
